// DistanceWeightedSampling_1752346657299
// MI455X (gfx1250) — matrix operand order repaired and checked by reading, not compiled, not run on hardware
//
#include <hip/hip_runtime.h>
#include <hip/hip_bf16.h>

// ---------------------------------------------------------------------------
// DistanceWeightedSampling, fused for MI455X (gfx1250, wave32, WMMA).
//
// Roofline: Gram matrix 2*B^2*D = 68.7 GFLOP via v_wmma_f32_16x16x32_f16
// (4.19M WMMA issues device-wide); HBM traffic ~437 MB (~19us floor at
// 23.3 TB/s). Scoring epilogue (sqrt + 4x v_log per element over 67M
// elements) is fused into the GEMM so the B^2 matrix never hits memory.
// Top-k ranking is invariant to the reference's row-constant shifts
// (global max, row-sum log), so scores = (mask ? log_w : -1e30) + gumbel.
// ---------------------------------------------------------------------------

#define BB 8192
#define DD 512
#define NPAIR (BB * 3)   // 24576 anchor/sample pairs

typedef __attribute__((ext_vector_type(16))) _Float16 v16h;
typedef __attribute__((ext_vector_type(8)))  _Float16 v8h;
typedef __attribute__((ext_vector_type(4)))  _Float16 v4h;
typedef __attribute__((ext_vector_type(8)))  float    v8f;

// A-matrix fragment (16x32 f16): lane L holds row M = L&15;
// elements 0..15 = K = (L>>4)*16 + {0..15}
// -> two adjacent 16-byte chunks.
__device__ __forceinline__ v16h make_afrag(const _Float16* p) {
    v8h lo = *(const v8h*)(p);
    v8h hi = *(const v8h*)(p + 8);
    v16h f;
#pragma unroll
    for (int e = 0; e < 8; ++e) { f[e] = lo[e]; f[8 + e] = hi[e]; }
    return f;
}

// B-matrix fragment (32x16 f16): lane L holds column N = L&15;
// elements 0..15 = K = (L>>4)*16 + {0..15}  -> one contiguous 32-byte chunk.
__device__ __forceinline__ v16h make_bfrag(const _Float16* p) {
    return *(const v16h*)p;
}

__device__ __forceinline__ void top3_insert(float s, int j,
                                            float& v0, int& i0,
                                            float& v1, int& i1,
                                            float& v2, int& i2) {
    if (s > v0)      { v2 = v1; i2 = i1; v1 = v0; i1 = i0; v0 = s; i0 = j; }
    else if (s > v1) { v2 = v1; i2 = i1; v1 = s;  i1 = j; }
    else if (s > v2) { v2 = s;  i2 = j; }
}

// distance-weighted sampling score; C = D = 512:
// log_w = (2-C)*log(d) - (C-3)/2 * log(1 - d^2/4)
__device__ __forceinline__ float dws_score(float dot, int i, int j,
                                           int ti, int tj, float u) {
    float d2   = 2.0f - 2.0f * dot + ((i == j) ? 1.0f : 0.0f);   // unit-norm rows
    float dist = fmaxf(sqrtf(fmaxf(d2, 0.0f)), 0.5f);
    float lw   = -510.0f * __logf(dist)
                 - 254.5f * __logf(fmaxf(1.0f - 0.25f * dist * dist, 1e-30f));
    float g    = -__logf(-__logf(u));                             // Gumbel
    bool  ok   = (ti != tj) && (dist < 1.4f);
    return (ok ? lw : -1e30f) + g;
}

// ---------------------------------------------------------------------------
// Kernel 1: f32 -> f16 embedding copy (workspace).
// ---------------------------------------------------------------------------
__global__ void dws_convert(const float* __restrict__ in,
                            _Float16* __restrict__ out) {
    int idx = blockIdx.x * blockDim.x + threadIdx.x;   // one float4 per thread
    float4 v = ((const float4*)in)[idx];
    v4h h;
    h[0] = (_Float16)v.x; h[1] = (_Float16)v.y;
    h[2] = (_Float16)v.z; h[3] = (_Float16)v.w;
    *(v4h*)(out + (size_t)idx * 4) = h;
}

// ---------------------------------------------------------------------------
// Kernel 2: fused Gram-matrix (WMMA) + score epilogue + per-row top-3.
// Grid: 512 blocks (16 rows each) x 128 threads (4 waves).
// Wave w covers columns [w*2048, w*2048+2048), 2 tiles (32 cols) in flight.
// ---------------------------------------------------------------------------
__global__ __launch_bounds__(128) void dws_score_topk(
    const _Float16* __restrict__ emb16,
    const int*      __restrict__ tgt,
    const float*    __restrict__ unif,
    int*            __restrict__ nidx) {
    __shared__ _Float16 sA[16 * DD];       // 16 KB A stripe
    __shared__ float    s_val[16][192];    // merge scratch
    __shared__ int      s_idx[16][192];

    const int tid     = threadIdx.x;
    const int wave    = tid >> 5;
    const int lane    = tid & 31;
    const int halfsel = lane >> 4;
    const int ncol    = lane & 15;
    const int rowbase = blockIdx.x * 16;

    // Stage the 16-row A stripe (contiguous 16 KB) into LDS.
    {
        const uint4* src = (const uint4*)(emb16 + (size_t)rowbase * DD);
        uint4*       dst = (uint4*)sA;
#pragma unroll
        for (int t = 0; t < 8; ++t) dst[tid + t * 128] = src[tid + t * 128];
    }
    __syncthreads();

    // Row classes for this lane's 8 C rows (M = v + 8*halfsel).
    int tiv[8];
#pragma unroll
    for (int v = 0; v < 8; ++v) tiv[v] = tgt[rowbase + v + 8 * halfsel];

    float t3v[8][3]; int t3i[8][3];
#pragma unroll
    for (int v = 0; v < 8; ++v)
#pragma unroll
        for (int c = 0; c < 3; ++c) { t3v[v][c] = -3.0e38f; t3i[v][c] = 0; }

    const int colstart = wave * 2048;
    const _Float16* pA = sA + (size_t)ncol * DD + halfsel * 16;

    for (int tp = 0; tp < 64; ++tp) {
        const int col0 = colstart + tp * 32;
        const int col1 = col0 + 16;
        const _Float16* gB0 = emb16 + (size_t)(col0 + ncol) * DD + halfsel * 16;
        const _Float16* gB1 = emb16 + (size_t)(col1 + ncol) * DD + halfsel * 16;

        v8f acc0 = {}; v8f acc1 = {};
#pragma unroll
        for (int ks = 0; ks < 16; ++ks) {
            v16h a  = make_afrag(pA  + ks * 32);
            v16h b0 = make_bfrag(gB0 + ks * 32);
            v16h b1 = make_bfrag(gB1 + ks * 32);
            acc0 = __builtin_amdgcn_wmma_f32_16x16x32_f16(
                false, a, false, b0, (short)0, acc0, false, false);
            acc1 = __builtin_amdgcn_wmma_f32_16x16x32_f16(
                false, a, false, b1, (short)0, acc1, false, false);
        }

        const int j0  = col0 + ncol;
        const int j1  = col1 + ncol;
        const int tj0 = tgt[j0];
        const int tj1 = tgt[j1];
#pragma unroll
        for (int v = 0; v < 8; ++v) {
            const int i = rowbase + v + 8 * halfsel;
            {
                float u = unif[(size_t)i * BB + j0];
                float s = dws_score(acc0[v], i, j0, tiv[v], tj0, u);
                top3_insert(s, j0, t3v[v][0], t3i[v][0],
                                   t3v[v][1], t3i[v][1],
                                   t3v[v][2], t3i[v][2]);
            }
            {
                float u = unif[(size_t)i * BB + j1];
                float s = dws_score(acc1[v], i, j1, tiv[v], tj1, u);
                top3_insert(s, j1, t3v[v][0], t3i[v][0],
                                   t3v[v][1], t3i[v][1],
                                   t3v[v][2], t3i[v][2]);
            }
        }
    }

    // Cross-lane / cross-wave merge: 192 candidates per row via LDS.
#pragma unroll
    for (int v = 0; v < 8; ++v) {
        const int m    = v + 8 * halfsel;
        const int slot = wave * 48 + ncol * 3;
#pragma unroll
        for (int c = 0; c < 3; ++c) {
            s_val[m][slot + c] = t3v[v][c];
            s_idx[m][slot + c] = t3i[v][c];
        }
    }
    __syncthreads();

    if (tid < 16) {
        float b0v = -3.0e38f, b1v = -3.0e38f, b2v = -3.0e38f;
        int   b0i = 0, b1i = 0, b2i = 0;
        for (int c = 0; c < 192; ++c)
            top3_insert(s_val[tid][c], s_idx[tid][c],
                        b0v, b0i, b1v, b1i, b2v, b2i);
        const int r = rowbase + tid;
        nidx[r * 3 + 0] = b0i;
        nidx[r * 3 + 1] = b1i;
        nidx[r * 3 + 2] = b2i;
    }
}

// ---------------------------------------------------------------------------
// Kernel 3: row gathers. out = [emb[a_idx] | emb[n_idx] | emb[p_idx]] flat.
// p_idx: other K-1 members of the contiguous group-of-4, ascending index
// (matches stable top_k on the equality mask with target = i // 4).
// ---------------------------------------------------------------------------
__global__ __launch_bounds__(128) void dws_gather(
    const float* __restrict__ emb,
    const int*   __restrict__ nidx,
    float*       __restrict__ out) {
    const int r   = blockIdx.x;      // 0 .. NPAIR-1
    const int i   = r / 3;
    const int c   = r - 3 * i;
    const int tid = threadIdx.x;

    const int q    = i & 3;
    const int base = i & ~3;
    const int rp   = base + c + ((c >= q) ? 1 : 0);
    const int rn   = nidx[r];

    const float4* ea = (const float4*)(emb + (size_t)i  * DD);
    const float4* en = (const float4*)(emb + (size_t)rn * DD);
    const float4* ep = (const float4*)(emb + (size_t)rp * DD);
    float4* o0 = (float4*)(out + (size_t)r * DD);
    float4* o1 = (float4*)(out + ((size_t)NPAIR + r) * DD);
    float4* o2 = (float4*)(out + ((size_t)2 * NPAIR + r) * DD);

    o0[tid] = ea[tid];   // 128 threads x float4 = 512 floats
    o1[tid] = en[tid];
    o2[tid] = ep[tid];
}

// ---------------------------------------------------------------------------
extern "C" void kernel_launch(void* const* d_in, const int* in_sizes, int n_in,
                              void* d_out, int out_size, void* d_ws,
                              size_t ws_size, hipStream_t stream) {
    const float* emb  = (const float*)d_in[0];   // [8192,512] f32
    const int*   tgt  = (const int*)d_in[1];     // [8192] i32
    const float* unif = (const float*)d_in[2];   // [8192,8192] f32
    float*       out  = (float*)d_out;           // 3 x [24576,512] f32

    _Float16* emb16 = (_Float16*)d_ws;                           // 8 MB
    int* nidx = (int*)((char*)d_ws + (size_t)BB * DD * 2);       // 96 KB

    // 1) f32 -> f16 embeddings (BB*DD/4 float4s / 256 threads)
    dws_convert<<<(BB * DD / 4) / 256, 256, 0, stream>>>(emb, emb16);

    // 2) fused WMMA Gram + score + per-row top-3 negatives
    dws_score_topk<<<BB / 16, 128, 0, stream>>>(emb16, tgt, unif, nidx);

    // 3) gather anchor / negative-sample / positive rows
    dws_gather<<<NPAIR, 128, 0, stream>>>(emb, nidx, out);
}
